// GIN_and_features_88089779241017
// MI455X (gfx1250) — compile-verified, matrix operand order checked by reading, not run on hardware
//
#include <hip/hip_runtime.h>
#include <hip/hip_bf16.h>

#define N_NODES  100000
#define N_EDGES  1600000
#define N_GRAPHS 512
#define D_HID    128
#define D_LAT    64
#define N_LAYERS 3
#define BN_EPS   1e-5f

typedef __attribute__((ext_vector_type(16))) _Float16 v16h;
typedef __attribute__((ext_vector_type(8)))  _Float16 v8h;
typedef __attribute__((ext_vector_type(8)))  float    v8f;

#define LDA 136  // 128 + 8 halves pad; row stride 272B (16B aligned, 4-bank skew)

__device__ __forceinline__ void atomic_add_f32(float* p, float v) {
    __hip_atomic_fetch_add(p, v, __ATOMIC_RELAXED, __HIP_MEMORY_SCOPE_AGENT);
}
__device__ __forceinline__ void atomic_add_lds(float* p, float v) {
    __hip_atomic_fetch_add(p, v, __ATOMIC_RELAXED, __HIP_MEMORY_SCOPE_WORKGROUP);
}

// ---------------- utility ----------------
__global__ void zero_f32(float* __restrict__ p, long n) {
    long i = (long)blockIdx.x * blockDim.x + threadIdx.x;
    long stride = (long)gridDim.x * blockDim.x;
    for (; i < n; i += stride) p[i] = 0.f;
}

// convW1[l][k][n] f32 -> W1t[l][n][k] f16 (all 3 layers)
__global__ void wt_conv(const float* __restrict__ W, _Float16* __restrict__ Wt) {
    int i = blockIdx.x * 256 + threadIdx.x;
    if (i >= N_LAYERS * D_HID * D_HID) return;
    int l = i >> 14, r = i & 16383, k = r >> 7, n = r & 127;
    Wt[(size_t)l * 16384 + (size_t)n * 128 + k] = (_Float16)W[i];
}

// ---------------- edge scatter-add (L2 resident) ----------------
__global__ __launch_bounds__(256)
void edge_agg(const float* __restrict__ x, const int* __restrict__ src,
              const int* __restrict__ dst, float* __restrict__ agg) {
    long gid = (long)blockIdx.x * 256 + threadIdx.x;
    int e = (int)(gid >> 5);
    if (e >= N_EDGES) return;
    int lane = (int)(gid & 31);
    int s = src[e], d = dst[e];
    float4 v = reinterpret_cast<const float4*>(x + (size_t)s * 128)[lane];
    float* ap = agg + (size_t)d * 128 + lane * 4;
    atomic_add_f32(ap + 0, v.x);
    atomic_add_f32(ap + 1, v.y);
    atomic_add_f32(ap + 2, v.z);
    atomic_add_f32(ap + 3, v.w);
}

// ---------------- GEMM1: H1 = leaky((x+agg) @ W1 + b1), + column stats ----------------
__global__ __launch_bounds__(256)
void gemm1_stats(const float* __restrict__ x, const float* __restrict__ agg,
                 const _Float16* __restrict__ Wt, const float* __restrict__ b1,
                 _Float16* __restrict__ H1,
                 float* __restrict__ colsum, float* __restrict__ colsq) {
    __shared__ __align__(16) _Float16 As[16 * LDA];
    __shared__ float s_sum[128], s_sq[128];
    const int t = threadIdx.x;
    const int row0 = blockIdx.x * 16;
    if (t < 128) { s_sum[t] = 0.f; s_sq[t] = 0.f; }
    {   // stage A panel (x+agg -> f16), thread t: row t>>4, 8 cols at (t&15)*8
        int r = t >> 4, k0 = (t & 15) * 8;
        const float4* xp = reinterpret_cast<const float4*>(x + (size_t)(row0 + r) * 128 + k0);
        const float4* ap = reinterpret_cast<const float4*>(agg + (size_t)(row0 + r) * 128 + k0);
        float4 x0 = xp[0], x1 = xp[1], a0 = ap[0], a1 = ap[1];
        _Float16* d = &As[r * LDA + k0];
        d[0] = (_Float16)(x0.x + a0.x); d[1] = (_Float16)(x0.y + a0.y);
        d[2] = (_Float16)(x0.z + a0.z); d[3] = (_Float16)(x0.w + a0.w);
        d[4] = (_Float16)(x1.x + a1.x); d[5] = (_Float16)(x1.y + a1.y);
        d[6] = (_Float16)(x1.z + a1.z); d[7] = (_Float16)(x1.w + a1.w);
    }
    __syncthreads();
    const int lane = t & 31;
    const int lm = lane & 15, g = lane >> 4;
    const int ntile = t >> 5;
    const int ncol = ntile * 16 + lm;

    v8f c = {0.f, 0.f, 0.f, 0.f, 0.f, 0.f, 0.f, 0.f};
    const _Float16* wr = Wt + (size_t)ncol * 128;
#pragma unroll
    for (int kk = 0; kk < 4; ++kk) {
        int kb = kk * 32 + g * 8;
        union AF { v16h v; v8h h[2]; } a, b;
        a.h[0] = *reinterpret_cast<const v8h*>(&As[lm * LDA + kb]);
        a.h[1] = *reinterpret_cast<const v8h*>(&As[lm * LDA + kb + 16]);
        b.h[0] = *reinterpret_cast<const v8h*>(wr + kb);
        b.h[1] = *reinterpret_cast<const v8h*>(wr + kb + 16);
        c = __builtin_amdgcn_wmma_f32_16x16x32_f16(false, a.v, false, b.v,
                                                   (short)0, c, false, false);
    }
    float bias = b1[ncol];
    float lsum = 0.f, lsq = 0.f;
#pragma unroll
    for (int r = 0; r < 8; ++r) {
        int row = row0 + g * 8 + r;
        float v = c[r] + bias;
        v = v > 0.f ? v : 0.2f * v;
        H1[(size_t)row * 128 + ncol] = (_Float16)v;
        lsum += v; lsq += v * v;
    }
    atomic_add_lds(&s_sum[ncol], lsum);
    atomic_add_lds(&s_sq[ncol], lsq);
    __syncthreads();
    if (t < 128) {
        atomic_add_f32(&colsum[t], s_sum[t]);
        atomic_add_f32(&colsq[t], s_sq[t]);
    }
}

// ---------------- fold BN into W2: W2t[n][k]=scale[k]*W2[k][n] (f16), b2eff ----------------
__global__ void fold_bn(const float* __restrict__ colsum, const float* __restrict__ colsq,
                        const float* __restrict__ g1, const float* __restrict__ bb1,
                        const float* __restrict__ W2, const float* __restrict__ b2,
                        _Float16* __restrict__ W2t, float* __restrict__ b2eff) {
    __shared__ float s_sc[128], s_sh[128];
    int t = threadIdx.x;  // 128 threads
    const float invM = 1.f / (float)N_NODES;
    float m = colsum[t] * invM;
    float var = colsq[t] * invM - m * m;
    float inv = rsqrtf(var + BN_EPS);
    float sc = g1[t] * inv;
    s_sc[t] = sc;
    s_sh[t] = bb1[t] - m * sc;
    __syncthreads();
    float acc = b2[t];
    for (int k = 0; k < 128; ++k) {
        float wv = W2[(size_t)k * 128 + t];
        W2t[(size_t)t * 128 + k] = (_Float16)(s_sc[k] * wv);
        acc += s_sh[k] * wv;
    }
    b2eff[t] = acc;
}

// ---------------- GEMM2: x = leaky(H1 @ W2eff + b2eff) ----------------
__global__ __launch_bounds__(256)
void gemm2(const _Float16* __restrict__ H1, const _Float16* __restrict__ W2t,
           const float* __restrict__ b2eff, float* __restrict__ xout) {
    __shared__ __align__(16) _Float16 As[16 * LDA];
    const int t = threadIdx.x;
    const int row0 = blockIdx.x * 16;
    {
        int r = t >> 4, k0 = (t & 15) * 8;
        v8h v = *reinterpret_cast<const v8h*>(H1 + (size_t)(row0 + r) * 128 + k0);
        *reinterpret_cast<v8h*>(&As[r * LDA + k0]) = v;
    }
    __syncthreads();
    const int lane = t & 31;
    const int lm = lane & 15, g = lane >> 4;
    const int ncol = (t >> 5) * 16 + lm;

    v8f c = {0.f, 0.f, 0.f, 0.f, 0.f, 0.f, 0.f, 0.f};
    const _Float16* wr = W2t + (size_t)ncol * 128;
#pragma unroll
    for (int kk = 0; kk < 4; ++kk) {
        int kb = kk * 32 + g * 8;
        union AF { v16h v; v8h h[2]; } a, b;
        a.h[0] = *reinterpret_cast<const v8h*>(&As[lm * LDA + kb]);
        a.h[1] = *reinterpret_cast<const v8h*>(&As[lm * LDA + kb + 16]);
        b.h[0] = *reinterpret_cast<const v8h*>(wr + kb);
        b.h[1] = *reinterpret_cast<const v8h*>(wr + kb + 16);
        c = __builtin_amdgcn_wmma_f32_16x16x32_f16(false, a.v, false, b.v,
                                                   (short)0, c, false, false);
    }
    float bias = b2eff[ncol];
#pragma unroll
    for (int r = 0; r < 8; ++r) {
        int row = row0 + g * 8 + r;
        float v = c[r] + bias;
        v = v > 0.f ? v : 0.2f * v;
        xout[(size_t)row * 128 + ncol] = v;
    }
}

// ---------------- pooling: pooled[batch[n]] += x[n] ----------------
__global__ __launch_bounds__(256)
void pool_sum(const float* __restrict__ x, const int* __restrict__ batch,
              float* __restrict__ pooled) {
    long gid = (long)blockIdx.x * 256 + threadIdx.x;
    int n = (int)(gid >> 5);
    if (n >= N_NODES) return;
    int lane = (int)(gid & 31);
    int b = batch[n];
    float4 v = reinterpret_cast<const float4*>(x + (size_t)n * 128)[lane];
    float* pp = pooled + (size_t)b * 128 + lane * 4;
    atomic_add_f32(pp + 0, v.x);
    atomic_add_f32(pp + 1, v.y);
    atomic_add_f32(pp + 2, v.z);
    atomic_add_f32(pp + 3, v.w);
}

// ---------------- head ----------------
__global__ void head_stats(const float* __restrict__ pooled, const float* __restrict__ g,
                           const float* __restrict__ b, float* __restrict__ pscale,
                           float* __restrict__ pshift) {
    int n = threadIdx.x;  // 128
    float s = 0.f, q = 0.f;
    for (int r = 0; r < N_GRAPHS; ++r) { float v = pooled[(size_t)r * 128 + n]; s += v; q += v * v; }
    float m = s * (1.f / N_GRAPHS);
    float var = q * (1.f / N_GRAPHS) - m * m;
    float inv = rsqrtf(var + BN_EPS);
    float sc = g[n] * inv;
    pscale[n] = sc; pshift[n] = b[n] - m * sc;
}

__global__ void cond_mlp(const float* __restrict__ feats, const float* __restrict__ W1,
                         const float* __restrict__ b1, const float* __restrict__ W2,
                         const float* __restrict__ b2, float* __restrict__ cat) {
    int gi = blockIdx.x * blockDim.x + threadIdx.x;
    if (gi >= N_GRAPHS) return;
    float f[7];
    for (int i = 0; i < 7; ++i) f[i] = feats[gi * 7 + i];
    float tv[8];
    for (int j = 0; j < 8; ++j) {
        float acc = b1[j];
        for (int i = 0; i < 7; ++i) acc += f[i] * W1[i * 8 + j];
        tv[j] = fmaxf(acc, 0.f);
    }
    for (int j = 0; j < 8; ++j) {
        float acc = b2[j];
        for (int i = 0; i < 8; ++i) acc += tv[i] * W2[i * 8 + j];
        cat[gi * 72 + 64 + j] = acc;
    }
}

__global__ void fc_pool(const float* __restrict__ pooled, const float* __restrict__ pscale,
                        const float* __restrict__ pshift, const float* __restrict__ fcW,
                        const float* __restrict__ fcb, float* __restrict__ cat) {
    int idx = blockIdx.x * 256 + threadIdx.x;  // 512*64
    int gi = idx >> 6, j = idx & 63;
    float acc = fcb[j];
    for (int k = 0; k < 128; ++k)
        acc += (pooled[(size_t)gi * 128 + k] * pscale[k] + pshift[k]) * fcW[k * 64 + j];
    cat[gi * 72 + j] = acc;
}

__global__ void f1_mlp(const float* __restrict__ cat, const float* __restrict__ W,
                       const float* __restrict__ b, float* __restrict__ h2) {
    int idx = blockIdx.x * 256 + threadIdx.x;
    int gi = idx >> 6, j = idx & 63;
    float acc = b[j];
    for (int k = 0; k < 72; ++k) acc += cat[gi * 72 + k] * W[k * 64 + j];
    h2[gi * 64 + j] = acc > 0.f ? acc : 0.2f * acc;
}

__global__ void stats2(const float* __restrict__ h2, const float* __restrict__ g,
                       const float* __restrict__ b, float* __restrict__ fs,
                       float* __restrict__ fh) {
    int n = threadIdx.x;  // 64
    float s = 0.f, q = 0.f;
    for (int r = 0; r < N_GRAPHS; ++r) { float v = h2[r * 64 + n]; s += v; q += v * v; }
    float m = s * (1.f / N_GRAPHS);
    float var = q * (1.f / N_GRAPHS) - m * m;
    float inv = rsqrtf(var + BN_EPS);
    float sc = g[n] * inv;
    fs[n] = sc; fh[n] = b[n] - m * sc;
}

__global__ void f2_mlp(const float* __restrict__ h2, const float* __restrict__ fs,
                       const float* __restrict__ fh, const float* __restrict__ W,
                       const float* __restrict__ b, float* __restrict__ out) {
    int idx = blockIdx.x * 256 + threadIdx.x;
    int gi = idx >> 6, j = idx & 63;
    float acc = b[j];
    for (int k = 0; k < 64; ++k) acc += (h2[gi * 64 + k] * fs[k] + fh[k]) * W[k * 64 + j];
    out[gi * 64 + j] = acc;
}

// ---------------- launch ----------------
extern "C" void kernel_launch(void* const* d_in, const int* in_sizes, int n_in,
                              void* d_out, int out_size, void* d_ws, size_t ws_size,
                              hipStream_t stream) {
    const float* x0     = (const float*)d_in[0];
    const int*   ei     = (const int*)d_in[1];
    const int*   src    = ei;
    const int*   dstp   = ei + N_EDGES;
    const int*   batch  = (const int*)d_in[2];
    const float* feats  = (const float*)d_in[3];
    const float* convW1 = (const float*)d_in[4];
    const float* convb1 = (const float*)d_in[5];
    const float* convg1 = (const float*)d_in[6];
    const float* convbb1= (const float*)d_in[7];
    const float* convW2 = (const float*)d_in[8];
    const float* convb2 = (const float*)d_in[9];
    const float* bn_g   = (const float*)d_in[10];
    const float* bn_b   = (const float*)d_in[11];
    const float* fcW    = (const float*)d_in[12];
    const float* fcb    = (const float*)d_in[13];
    const float* condW1 = (const float*)d_in[14];
    const float* condb1 = (const float*)d_in[15];
    const float* condW2 = (const float*)d_in[16];
    const float* condb2 = (const float*)d_in[17];
    const float* fW1    = (const float*)d_in[18];
    const float* fb1    = (const float*)d_in[19];
    const float* f_g    = (const float*)d_in[20];
    const float* f_b    = (const float*)d_in[21];
    const float* fW2    = (const float*)d_in[22];
    const float* fb2    = (const float*)d_in[23];
    float* out = (float*)d_out;

    char* ws = (char*)d_ws;
    size_t off = 0;
    auto alloc = [&](size_t bytes) -> void* {
        void* p = ws + off;
        off += (bytes + 255) & ~(size_t)255;
        return p;
    };
    float*     agg    = (float*)alloc((size_t)N_NODES * 128 * 4);
    float*     x_buf  = (float*)alloc((size_t)N_NODES * 128 * 4);
    _Float16*  H1     = (_Float16*)alloc((size_t)N_NODES * 128 * 2);
    _Float16*  W1t    = (_Float16*)alloc((size_t)N_LAYERS * 16384 * 2);
    _Float16*  W2t    = (_Float16*)alloc((size_t)16384 * 2);
    float*     b2eff  = (float*)alloc(128 * 4);
    float*     stats  = (float*)alloc(256 * 4);   // colsum[128] | colsq[128]
    float*     pooled = (float*)alloc((size_t)N_GRAPHS * 128 * 4);
    float*     pscale = (float*)alloc(128 * 4);
    float*     pshift = (float*)alloc(128 * 4);
    float*     catb   = (float*)alloc((size_t)N_GRAPHS * 72 * 4);
    float*     h2     = (float*)alloc((size_t)N_GRAPHS * 64 * 4);
    float*     fs     = (float*)alloc(64 * 4);
    float*     fh     = (float*)alloc(64 * 4);
    float*     colsum = stats;
    float*     colsq  = stats + 128;

    wt_conv<<<(N_LAYERS * 16384 + 255) / 256, 256, 0, stream>>>(convW1, W1t);
    zero_f32<<<256, 256, 0, stream>>>(pooled, (long)N_GRAPHS * 128);

    for (int l = 0; l < N_LAYERS; ++l) {
        const float* xin = (l == 0) ? x0 : x_buf;
        zero_f32<<<2048, 256, 0, stream>>>(agg, (long)N_NODES * 128);
        zero_f32<<<1, 256, 0, stream>>>(stats, 256);
        edge_agg<<<(N_EDGES * 32) / 256, 256, 0, stream>>>(xin, src, dstp, agg);
        gemm1_stats<<<N_NODES / 16, 256, 0, stream>>>(
            xin, agg, W1t + (size_t)l * 16384, convb1 + l * 128, H1, colsum, colsq);
        fold_bn<<<1, 128, 0, stream>>>(colsum, colsq, convg1 + l * 128, convbb1 + l * 128,
                                       convW2 + (size_t)l * 16384, convb2 + l * 128,
                                       W2t, b2eff);
        gemm2<<<N_NODES / 16, 256, 0, stream>>>(H1, W2t, b2eff, x_buf);
    }

    pool_sum<<<(N_NODES * 32) / 256, 256, 0, stream>>>(x_buf, batch, pooled);
    head_stats<<<1, 128, 0, stream>>>(pooled, bn_g, bn_b, pscale, pshift);
    cond_mlp<<<2, 256, 0, stream>>>(feats, condW1, condb1, condW2, condb2, catb);
    fc_pool<<<(N_GRAPHS * 64) / 256, 256, 0, stream>>>(pooled, pscale, pshift, fcW, fcb, catb);
    f1_mlp<<<(N_GRAPHS * 64) / 256, 256, 0, stream>>>(catb, fW1, fb1, h2);
    stats2<<<1, 64, 0, stream>>>(h2, f_g, f_b, fs, fh);
    f2_mlp<<<(N_GRAPHS * 64) / 256, 256, 0, stream>>>(h2, fs, fh, fW2, fb2, out);
}
